// general_RNN_44555990729261
// MI455X (gfx1250) — matrix operand order repaired and checked by reading, not compiled, not run on hardware
//
#include <hip/hip_runtime.h>
#include <math.h>

typedef __attribute__((ext_vector_type(16))) _Float16 v16h;
typedef __attribute__((ext_vector_type(8)))  _Float16 v8h;
typedef __attribute__((ext_vector_type(8)))  float    v8f;

#define VIS 128
#define FULL 512
#define TSTEPS 2048

// Branch-free tanh via raw v_exp_f32 (base-2) + v_rcp_f32.
__device__ __forceinline__ float tanh_fast(float x) {
    float e = __builtin_amdgcn_exp2f(x * 2.8853900817779268f); // 2*log2(e)
    float r = __builtin_amdgcn_rcpf(1.0f + e);
    return 1.0f - 2.0f * r;
}

// ---------------------------------------------------------------------------
// Kernel 1: swizzle W (f32 row-major [n][k]) into f16 WMMA B-fragment order:
// Bfrag[n_tile][k_step][lane][h], lane L: n = n_tile*16 + (L%16),
// k = k_step*32 + (L/16)*16 + h, value = W[n][k].
// Fragment (n_tile, k_step) element index = n_tile*8192 + k_step*512 + lane*16 + h.
// ---------------------------------------------------------------------------
__global__ void wfrag_kernel(const float* __restrict__ W,
                             _Float16* __restrict__ Bfrag) {
    int idx = blockIdx.x * blockDim.x + threadIdx.x;
    if (idx >= FULL * FULL) return;
    int n = idx >> 9;
    int k = idx & 511;
    int n_tile  = n >> 4;
    int lane_n  = n & 15;
    int k_step  = k >> 5;
    int lane_hi = (k >> 4) & 1;
    int h       = k & 15;
    int lane    = lane_hi * 16 + lane_n;
    int dst = n_tile * 8192 + k_step * 512 + lane * 16 + h;
    Bfrag[dst] = (_Float16)W[n * FULL + k];
}

// ---------------------------------------------------------------------------
// Kernel 2: persistent single-workgroup recurrence.
//  - R double-buffered in dynamic LDS as f16 (256KB < 320KB WGP LDS).
//  - 32 waves = 4 m-groups x 8 n-groups; each wave: 2 m-tiles x 4 n-tiles,
//    8 v8f accumulators, rolled K-loop of 16 x (8 x v_wmma_f32_16x16x32_f16)
//    with loop-carried pointers so the loads stay one clause + immediates.
//  - Final timestep peeled: diag(U) = WR + b (Lambda==1 on the diagonal).
// ---------------------------------------------------------------------------
__global__ __launch_bounds__(1024, 1)
void rnn_kernel(const float* __restrict__ X,
                const float* __restrict__ b,
                const _Float16* __restrict__ Bfrag,
                float* __restrict__ out) {
    extern __shared__ _Float16 smem[];          // 2 * 128*512 halves

    const int tid     = threadIdx.x;
    const int lane    = tid & 31;
    const int wave    = tid >> 5;               // 0..31
    const int ng      = wave & 7;               // 4 n-tiles each
    const int mg      = wave >> 3;              // 2 m-tiles each
    const int lane_lo = lane & 15;
    const int lane_hi = lane >> 4;
    const int klo     = lane_hi * 16;
    const int mrow    = mg * 32 + lane_hi * 8;  // first output row (r=0,mi=0)

    // Loop-invariant fragment base pointers / offsets.
    const _Float16* bfrag0 = Bfrag + (ng * 4) * 8192 + lane * 16;
    const int a_base = (mg * 32 + lane_lo) * FULL + klo; // row m (mi=0), k=klo

    int   nn[4];
    float bn[4];
#pragma unroll
    for (int j = 0; j < 4; ++j) {
        nn[j] = (ng * 4 + j) * 16 + lane_lo;
        bn[j] = b[nn[j]];
    }

    // Zero R buffer 0 (R0 = 0).
    {
        unsigned int* z = (unsigned int*)smem;
        for (int i = tid; i < (VIS * FULL) / 2; i += 1024) z[i] = 0u;
    }
    __syncthreads();

    // K-reduction GEMM: acc[mi][j] += A(R)[m-tile mi] x B(W^T)[n-tile j].
    // Rolled loop + pointer chase => clean clauses, no WMMA/VALU reg overlap.
    auto compute_acc = [&](int rc_off, v8f (&acc)[2][4]) {
        const _Float16* bp = bfrag0;
        const _Float16* ap = smem + rc_off + a_base;
#pragma unroll 1
        for (int ks = 0; ks < 16; ++ks) {
            v16h bf[4];
#pragma unroll
            for (int j = 0; j < 4; ++j)
                bf[j] = *(const v16h*)(bp + j * 8192);

            v16h a[2];
#pragma unroll
            for (int mi = 0; mi < 2; ++mi) {
                v8h lo = *(const v8h*)(ap + mi * 8192);
                v8h hi = *(const v8h*)(ap + mi * 8192 + 8);
#pragma unroll
                for (int i = 0; i < 8; ++i) {
                    a[mi][i] = lo[i]; a[mi][i + 8] = hi[i];
                }
            }

#pragma unroll
            for (int mi = 0; mi < 2; ++mi)
#pragma unroll
                for (int j = 0; j < 4; ++j)
                    acc[mi][j] = __builtin_amdgcn_wmma_f32_16x16x32_f16(
                        false, a[mi], false, bf[j], (short)0, acc[mi][j],
                        false, false);

            bp += 512;   // next k_step fragment block
            ap += 32;    // next k chunk of R row
        }
    };

    const v8f zero = {};
    int cur = 0;

    // ---- main loop: steps 0 .. TSTEPS-2 (update R, no diag extraction) ----
    for (int t = 0; t < TSTEPS - 1; ++t) {
        const int rc_off = cur ? VIS * FULL : 0;
        const int rn_off = cur ? 0 : VIS * FULL;

        float xh[4];
#pragma unroll
        for (int j = 0; j < 4; ++j)
            xh[j] = (nn[j] < VIS) ? 0.5f * X[t * VIS + nn[j]] : 0.0f;

        v8f acc[2][4];
#pragma unroll
        for (int mi = 0; mi < 2; ++mi)
#pragma unroll
            for (int j = 0; j < 4; ++j) acc[mi][j] = zero;

        compute_acc(rc_off, acc);

        // Epilogue: WR = acc + b[n];
        // U = (n<VIS && m!=n) ? 0.5*WR + 0.5*x : WR;  R_new = tanh(U).
        // C/D layout: VGPR r -> M = r + 8*(lane/16), N = lane%16.
#pragma unroll
        for (int j = 0; j < 4; ++j) {
            const int n    = nn[j];
            const int base = rn_off + mrow * FULL + n;   // one add per j
#pragma unroll
            for (int mi = 0; mi < 2; ++mi) {
#pragma unroll
                for (int r = 0; r < 8; ++r) {
                    int m = mrow + mi * 16 + r;
                    float wr = acc[mi][j][r] + bn[j];
                    float U  = (n < VIS && m != n) ? fmaf(0.5f, wr, xh[j])
                                                   : wr;
                    // constant (mi*16+r)*FULL folds into ds_store offset
                    smem[base + (mi * 16 + r) * FULL] =
                        (_Float16)tanh_fast(U);
                }
            }
        }
        __syncthreads();
        cur ^= 1;
    }

    // ---- peeled final step: only diag(U) needed; on the diagonal Lambda==1
    //      so U = WR + b. No tanh, no R store. ----
    {
        const int rc_off = cur ? VIS * FULL : 0;
        v8f acc[2][4];
#pragma unroll
        for (int mi = 0; mi < 2; ++mi)
#pragma unroll
            for (int j = 0; j < 4; ++j) acc[mi][j] = zero;

        compute_acc(rc_off, acc);

#pragma unroll
        for (int mi = 0; mi < 2; ++mi)
#pragma unroll
            for (int j = 0; j < 4; ++j)
#pragma unroll
                for (int r = 0; r < 8; ++r) {
                    int m = mrow + mi * 16 + r;
                    int n = nn[j];
                    if (n < VIS && m == n) out[m] = acc[mi][j][r] + bn[j];
                }
    }
}

// ---------------------------------------------------------------------------
// Launcher: inputs (X [2048*128] f32, W [512*512] f32, b [512] f32),
// output = 128 f32. d_ws holds the f16-swizzled W (512 KB).
// ---------------------------------------------------------------------------
extern "C" void kernel_launch(void* const* d_in, const int* in_sizes, int n_in,
                              void* d_out, int out_size, void* d_ws,
                              size_t ws_size, hipStream_t stream) {
    const float* X = (const float*)d_in[0];
    const float* W = (const float*)d_in[1];
    const float* b = (const float*)d_in[2];
    float* out = (float*)d_out;
    _Float16* Bfrag = (_Float16*)d_ws;

    (void)in_sizes; (void)n_in; (void)out_size; (void)ws_size;

    wfrag_kernel<<<(FULL * FULL + 255) / 256, 256, 0, stream>>>(W, Bfrag);

    size_t lds_bytes = 2u * VIS * FULL * sizeof(_Float16);
    rnn_kernel<<<1, 1024, lds_bytes, stream>>>(X, b, Bfrag, out);
}
